// W4A8B8O8Linear_40243843564265
// MI455X (gfx1250) — 8-bit matrix operand order repaired and checked by reading, compiled, not run on hardware
//
#include <hip/hip_runtime.h>
#include <hip/hip_fp16.h>

typedef __attribute__((ext_vector_type(8))) int          v8i;
typedef __attribute__((ext_vector_type(4))) int          i4v;
typedef __attribute__((ext_vector_type(2))) unsigned int u2v;
typedef __attribute__((address_space(3))) signed char    lds_char;

union Frag {
  v8i v;
  i4v q[2];
  u2v d[4];
};

static constexpr int Mdim  = 8192;
static constexpr int Kdim  = 4096;
static constexpr int Ndim  = 11008;
static constexpr int GRP   = 128;
static constexpr int NGRP  = Kdim / GRP;

__global__ void pack_x_kernel(const int* __restrict__ x, int* __restrict__ xp4) {
  const size_t i = (size_t)blockIdx.x * blockDim.x + threadIdx.x;
  const i4v v = ((const i4v*)x)[i];
  unsigned p = (unsigned)(v[0] & 0xff)
             | ((unsigned)(v[1] & 0xff) << 8)
             | ((unsigned)(v[2] & 0xff) << 16)
             | ((unsigned)(v[3] & 0xff) << 24);
  xp4[i] = (int)p;
}

__global__ void pack_w_kernel(const int* __restrict__ qw, signed char* __restrict__ wt) {
  __shared__ signed char tile[32][33];
  const int t  = threadIdx.x;
  const int n0 = blockIdx.x * 32;
  const int k0 = blockIdx.y * 32;

  const int nl = t & 31;
#pragma unroll
  for (int i = 0; i < 4; ++i) {
    const int kl = (t >> 5) * 4 + i;
    tile[kl][nl] = (signed char)(qw[(size_t)(k0 + kl) * Ndim + (n0 + nl)] - 8);
  }
  __syncthreads();

  const int nr = t >> 3;
  const int cw = t & 7;
  unsigned p = 0;
#pragma unroll
  for (int b = 0; b < 4; ++b)
    p |= ((unsigned)(unsigned char)tile[cw * 4 + b][nr]) << (8 * b);
  *(int*)(wt + (size_t)(n0 + nr) * Kdim + (size_t)(k0 + cw * 4)) = (int)p;
}

__global__ __launch_bounds__(256)
void w4a8_gemm_kernel(const signed char* __restrict__ xp,
                      const signed char* __restrict__ wt,
                      const float*  __restrict__ input_scale,
                      const float*  __restrict__ s_channel,
                      const __half* __restrict__ s_group,
                      float* __restrict__ out) {
  __shared__ __align__(16) signed char As[2][64  * GRP];
  __shared__ __align__(16) signed char Bs[2][128 * GRP];

  const int tid  = threadIdx.x;
  const int lane = tid & 31;
  const int wave = tid >> 5;
  const int wm   = wave >> 2;
  const int wn   = wave & 3;
  const int l15  = lane & 15;
  const int sel  = lane >> 4;

  const int n0 = blockIdx.x * 128;
  const int m0 = blockIdx.y * 64;

  auto issue_async = [&](int g, int p) {
    const size_t kb = (size_t)g * GRP;
#pragma unroll
    for (int j = 0; j < 2; ++j) {
      const int c = tid + j * 256;
      const signed char* gp =
          xp + (size_t)(m0 + (c >> 3)) * Kdim + kb + (size_t)(c & 7) * 16;
      const unsigned la = (unsigned)(size_t)(lds_char*)&As[p][c * 16];
      asm volatile("global_load_async_to_lds_b128 %0, %1, off"
                   :: "v"(la), "v"((unsigned long long)(size_t)gp) : "memory");
    }
#pragma unroll
    for (int j = 0; j < 4; ++j) {
      const int c = tid + j * 256;
      const signed char* gp =
          wt + (size_t)(n0 + (c >> 3)) * Kdim + kb + (size_t)(c & 7) * 16;
      const unsigned la = (unsigned)(size_t)(lds_char*)&Bs[p][c * 16];
      asm volatile("global_load_async_to_lds_b128 %0, %1, off"
                   :: "v"(la), "v"((unsigned long long)(size_t)gp) : "memory");
    }
    if (g + 1 < NGRP) {
      const size_t kb2 = kb + GRP;
      __builtin_prefetch(xp + (size_t)(m0 + (tid >> 3)) * Kdim + kb2 + (size_t)(tid & 7) * 16, 0, 1);
      __builtin_prefetch(wt + (size_t)(n0 + (tid >> 3)) * Kdim + kb2 + (size_t)(tid & 7) * 16, 0, 1);
    }
  };
  auto wait_async = [&]() {
    asm volatile("s_wait_asynccnt 0" ::: "memory");
  };

  const v8i vzero = {0, 0, 0, 0, 0, 0, 0, 0};

  auto compute_group = [&](int p, v8i (&c)[2][2]) {
    const signed char* as = As[p];
    const signed char* bs = Bs[p];
#pragma unroll
    for (int kk = 0; kk < GRP; kk += 64) {
      Frag fa[2], fb[2];
#pragma unroll
      for (int mt = 0; mt < 2; ++mt) {
        const int base = (wm * 32 + mt * 16 + l15) * GRP + kk + sel * 16;
#pragma unroll
        for (int cc = 0; cc < 4; ++cc)
          fa[mt].d[cc] = *(const u2v*)(as + base + (cc & 1) * 8 + (cc >> 1) * 32);
      }
#pragma unroll
      for (int nt = 0; nt < 2; ++nt) {
        const int base = (wn * 32 + nt * 16 + l15) * GRP + kk + sel * 16;
        fb[nt].q[0] = *(const i4v*)(bs + base);
        fb[nt].q[1] = *(const i4v*)(bs + base + 32);
      }
#pragma unroll
      for (int mt = 0; mt < 2; ++mt)
#pragma unroll
        for (int nt = 0; nt < 2; ++nt)
          c[mt][nt] = __builtin_amdgcn_wmma_i32_16x16x64_iu8(
               false, fa[mt].v,
               true,  fb[nt].v,
              (kk == 0) ? vzero : c[mt][nt],
               false,  false);
    }
  };

  float accF[2][2][8];
#pragma unroll
  for (int mt = 0; mt < 2; ++mt)
#pragma unroll
    for (int nt = 0; nt < 2; ++nt)
#pragma unroll
      for (int r = 0; r < 8; ++r) accF[mt][nt][r] = 0.0f;

  auto load_scales = [&](int g, float (&sg)[2]) {
#pragma unroll
    for (int nt = 0; nt < 2; ++nt) {
      const int ncol = n0 + wn * 32 + nt * 16 + l15;
      sg[nt] = __half2float(s_group[(size_t)g * Ndim + ncol]);
    }
  };
  auto dequant = [&](const v8i (&c)[2][2], const float (&sg)[2]) {
#pragma unroll
    for (int nt = 0; nt < 2; ++nt)
#pragma unroll
      for (int mt = 0; mt < 2; ++mt)
#pragma unroll
        for (int r = 0; r < 8; ++r)
          accF[mt][nt][r] += (float)c[mt][nt][r] * sg[nt];
  };

  v8i   cA[2][2], cB[2][2];
  float sgA[2], sgB[2];

  issue_async(0, 0);
  wait_async();
  __syncthreads();
  issue_async(1, 1);
  load_scales(0, sgA);
  compute_group(0, cA);

  for (int g = 1; g + 1 < NGRP; g += 2) {
    wait_async();
    __syncthreads();
    issue_async(g + 1, 0);
    load_scales(g, sgB);
    compute_group(1, cB);
    dequant(cA, sgA);

    wait_async();
    __syncthreads();
    if (g + 2 < NGRP) issue_async(g + 2, 1);
    load_scales(g + 1, sgA);
    compute_group(0, cA);
    dequant(cB, sgB);
  }

  wait_async();
  __syncthreads();
  load_scales(NGRP - 1, sgB);
  compute_group(1, cB);
  dequant(cA, sgA);
  dequant(cB, sgB);

  const float is0 = input_scale[0];
#pragma unroll
  for (int nt = 0; nt < 2; ++nt) {
    const int n = n0 + wn * 32 + nt * 16 + l15;
    const float sc = is0 * s_channel[n];
#pragma unroll
    for (int mt = 0; mt < 2; ++mt) {
      const int mbase = m0 + wm * 32 + mt * 16 + sel * 8;
#pragma unroll
      for (int r = 0; r < 8; ++r) {
        float v = accF[mt][nt][r] * sc;
        v = fminf(fmaxf(v, -128.0f), 127.0f);
        out[(size_t)(mbase + r) * Ndim + n] = v;
      }
    }
  }
}

extern "C" void kernel_launch(void* const* d_in, const int* in_sizes, int n_in,
                              void* d_out, int out_size, void* d_ws, size_t ws_size,
                              hipStream_t stream) {
  const int*    x   = (const int*)d_in[0];
  const int*    qw  = (const int*)d_in[1];
  const float*  is  = (const float*)d_in[2];
  const float*  sch = (const float*)d_in[3];
  const __half* sg  = (const __half*)d_in[4];
  float* out = (float*)d_out;

  signed char* xp = (signed char*)d_ws;
  signed char* wt = xp + (size_t)Mdim * Kdim;

  const int xblocks = (int)(((size_t)Mdim * Kdim / 4) / 256);
  pack_x_kernel<<<xblocks, 256, 0, stream>>>(x, (int*)xp);
  pack_w_kernel<<<dim3(Ndim / 32, Kdim / 32), 256, 0, stream>>>(qw, wt);
  w4a8_gemm_kernel<<<dim3(Ndim / 128, Mdim / 64), 256, 0, stream>>>(
      xp, wt, is, sch, sg, out);
}
